// SABlock_73839077753151
// MI455X (gfx1250) — matrix operand order repaired and checked by reading, not compiled, not run on hardware
//
#include <hip/hip_runtime.h>
#include <hip/hip_bf16.h>

// ---------------------------------------------------------------------------
// BoTNet SA block for MI455X (gfx1250, wave32, WMMA bf16 + TDM staging)
//   1) BN1+ReLU, pack x -> xb bf16 [b][t][c]
//   2) convert weights to bf16
//   3) QKV GEMM: TDM-staged A tile in LDS, dual N-tile WMMA pipelines
//   4) q x rel GEMMs (WMMA)
//   5) flash attention: TDM-staged K/V in LDS, WMMA QK^T + online softmax +
//      WMMA PV (P transposed through LDS)
//   6) residual + BN2+ReLU
//   7) 3x3 conv implicit GEMM: LDS halo (border-free inner loop), WMMA
// ---------------------------------------------------------------------------

#define BATCH 8
#define CH    512
#define HHDIM 32
#define WWDIM 32
#define LTOK  1024
#define HEADS 4
#define DHEAD 128
#define SCALE 0.08838834764831845f   // 128^-0.5
#define EPS   1e-5f

typedef __attribute__((ext_vector_type(16))) __bf16 v16bf;
typedef __attribute__((ext_vector_type(8)))  __bf16 v8bf;
typedef __attribute__((ext_vector_type(8)))  float  v8f;
typedef __attribute__((ext_vector_type(4)))  unsigned int u32x4;
typedef __attribute__((ext_vector_type(8)))  int i32x8;
typedef __attribute__((ext_vector_type(4)))  int i32x4;

#if __has_builtin(__builtin_amdgcn_tensor_load_to_lds)
#define HAVE_TDM 1
#else
#define HAVE_TDM 0
#endif

// ---------------- WMMA fragment helpers (ISA 7.12.2 layouts) ----------------

__device__ __forceinline__ v16bf load_a_frag(const __bf16* base, int ld, int kbase) {
    const int lane = threadIdx.x & 31;
    const int m = lane & 15, half = lane >> 4;
    const __bf16* p = base + (size_t)m * ld + kbase + half * 8;
    v8bf lo = *(const v8bf*)p;
    v8bf hi = *(const v8bf*)(p + 16);
    return __builtin_shufflevector(lo, hi, 0,1,2,3,4,5,6,7,8,9,10,11,12,13,14,15);
}

// A fragment from a per-lane row pointer (row already selected by caller).
__device__ __forceinline__ v16bf load_a_frag_rowptr(const __bf16* rowp, int kbase) {
    const int half = (threadIdx.x >> 4) & 1;
    const __bf16* p = rowp + kbase + half * 8;
    v8bf lo = *(const v8bf*)p;
    v8bf hi = *(const v8bf*)(p + 16);
    return __builtin_shufflevector(lo, hi, 0,1,2,3,4,5,6,7,8,9,10,11,12,13,14,15);
}

__device__ __forceinline__ v16bf load_b_frag(const __bf16* base, int ld, int kbase) {
    const int lane = threadIdx.x & 31;
    const int n = lane & 15, half = lane >> 4;
    const __bf16* p = base + (size_t)n * ld + kbase + half * 8;
    v8bf lo = *(const v8bf*)p;
    v8bf hi = *(const v8bf*)(p + 16);
    return __builtin_shufflevector(lo, hi, 0,1,2,3,4,5,6,7,8,9,10,11,12,13,14,15);
}

__device__ __forceinline__ v8f wmma_bf16(v16bf a, v16bf b, v8f c) {
    return __builtin_amdgcn_wmma_f32_16x16x32_bf16(false, a, false, b, (short)0, c, false, false);
}

// ---------------- TDM: 2D tile load Global -> LDS (D# per ISA ch.8) ---------

#if HAVE_TDM
__device__ __forceinline__ void tdm_load_2d(unsigned lds_addr, unsigned long long gaddr,
                                            unsigned dim0, unsigned dim1,
                                            unsigned long long stride0,
                                            unsigned tile0, unsigned tile1) {
    u32x4 g0;
    g0[0] = 1u;                                        // count=1, user mode
    g0[1] = lds_addr;                                  // LDS byte address
    g0[2] = (unsigned)(gaddr & 0xFFFFFFFFull);         // global_addr[31:0]
    g0[3] = (unsigned)((gaddr >> 32) & 0x01FFFFFFull) | (2u << 30);  // addr[56:32] | type=2
    i32x8 g1;
    g1[0] = (int)(1u << 16);                           // data_size=1 (2 bytes/elem)
    g1[1] = (int)((dim0 & 0xFFFFu) << 16);             // tensor_dim0[15:0]
    g1[2] = (int)(((dim0 >> 16) & 0xFFFFu) | ((dim1 & 0xFFFFu) << 16));
    g1[3] = (int)(((dim1 >> 16) & 0xFFFFu) | ((tile0 & 0xFFFFu) << 16));
    g1[4] = (int)(tile1 & 0xFFFFu);                    // tile_dim1; tile_dim2=0 (2D)
    g1[5] = (int)(unsigned)(stride0 & 0xFFFFFFFFull);  // tensor_dim0_stride[31:0]
    g1[6] = (int)(unsigned)((stride0 >> 32) & 0xFFFFull);
    g1[7] = 0;
    i32x4 z4 = {};
#if defined(__clang_major__) && (__clang_major__ >= 23)
    i32x8 z8 = {};
    __builtin_amdgcn_tensor_load_to_lds(g0, g1, z4, z4, z8, 0);
#else
    __builtin_amdgcn_tensor_load_to_lds(g0, g1, z4, z4, 0);
#endif
}

__device__ __forceinline__ void tdm_wait0() {
#if __has_builtin(__builtin_amdgcn_s_wait_tensorcnt)
    __builtin_amdgcn_s_wait_tensorcnt(0);
#else
    asm volatile("s_wait_tensorcnt 0x0" ::: "memory");
#endif
}
#endif

// ---------------- 1) BN1 + ReLU + NHWC bf16 pack ----------------

__global__ void k_bn1_pack(const float* __restrict__ x,
                           const float* __restrict__ g, const float* __restrict__ be,
                           const float* __restrict__ mu, const float* __restrict__ va,
                           __bf16* __restrict__ xb) {
    int idx = blockIdx.x * blockDim.x + threadIdx.x;
    if (idx >= BATCH * CH * LTOK) return;
    int t = idx % LTOK;
    int c = (idx / LTOK) % CH;
    int b = idx / (CH * LTOK);
    float inv = g[c] * __frsqrt_rn(va[c] + EPS);
    float beta = be[c] - mu[c] * inv;
    float v = fmaxf(x[idx] * inv + beta, 0.f);
    xb[((size_t)b * LTOK + t) * CH + c] = (__bf16)v;
}

// ---------------- 2) weight conversions ----------------

__global__ void k_cvt_bf16(const float* __restrict__ in, __bf16* __restrict__ out, int n) {
    int i = blockIdx.x * blockDim.x + threadIdx.x;
    if (i < n) out[i] = (__bf16)in[i];
}

__global__ void k_cvt_rel(const float* __restrict__ rh, const float* __restrict__ rw,
                          __bf16* __restrict__ rhp, __bf16* __restrict__ rwp) {
    int i = blockIdx.x * blockDim.x + threadIdx.x;
    if (i >= 64 * 128) return;
    int r = i / 128, d = i % 128;
    rhp[i] = (__bf16)((r < 63) ? rh[r * 128 + d] : 0.f);
    rwp[i] = (__bf16)((r < 63) ? rw[r * 128 + d] : 0.f);
}

__global__ void k_cvt_wfc(const float* __restrict__ wfc, __bf16* __restrict__ wfcT) {
    int i = blockIdx.x * blockDim.x + threadIdx.x;
    if (i >= 9 * CH * CH) return;
    int c = i % CH;
    int o = (i / CH) % CH;
    int tap = i / (CH * CH);
    int ky = tap / 3, kx = tap % 3;
    wfcT[i] = (__bf16)wfc[(((size_t)o * CH + c) * 3 + ky) * 3 + kx];
}

// ---------------- 3) QKV GEMM: LDS A tile + dual N-tile pipelines -----------
// grid (64, 8), block 128 ; A tile (16x512) shared by all 4 waves via TDM.

__global__ void __launch_bounds__(128) k_qkv(const __bf16* __restrict__ xb,
                                             const __bf16* __restrict__ wqk,
                                             const __bf16* __restrict__ wv,
                                             __bf16* __restrict__ qbf,
                                             __bf16* __restrict__ kbf,
                                             __bf16* __restrict__ vt) {
    const int mt = blockIdx.x, b = blockIdx.y;
    const int wave = threadIdx.x >> 5, lane = threadIdx.x & 31;
    const int half = lane >> 4, nl = lane & 15;

    __shared__ __bf16 At[16][CH];   // 16 KB, contiguous slab of xb

    const __bf16* ap = xb + ((size_t)b * LTOK + mt * 16) * CH;
#if HAVE_TDM
    if (wave == 0) {
        tdm_load_2d((unsigned)(size_t)&At[0][0], (unsigned long long)(size_t)ap,
                    CH, 16, CH, CH, 16);
        tdm_wait0();
    }
#else
    for (int i = threadIdx.x; i < 16 * CH / 8; i += 128)
        ((v8bf*)&At[0][0])[i] = *(const v8bf*)(ap + i * 8);
#endif
    __syncthreads();

    // each wave: 24 N-tiles, processed 2 at a time (independent WMMA chains)
    for (int np = 0; np < 12; ++np) {
        const int ob0 = (wave * 24 + np * 2) * 16;
        const int ob1 = ob0 + 16;
        const __bf16* w0 = (ob0 < 1024) ? (wqk + (size_t)ob0 * CH)
                                        : (wv + (size_t)(ob0 - 1024) * CH);
        const __bf16* w1 = (ob1 < 1024) ? (wqk + (size_t)ob1 * CH)
                                        : (wv + (size_t)(ob1 - 1024) * CH);
        v8f acc0 = {}, acc1 = {};
        v16bf b0c = load_b_frag(w0, CH, 0);
        v16bf b1c = load_b_frag(w1, CH, 0);
#pragma unroll
        for (int kk = 0; kk < 16; ++kk) {
            v16bf a = load_a_frag(&At[0][0], CH, kk * 32);   // LDS
            v16bf b0n = b0c, b1n = b1c;
            if (kk < 15) {
                b0n = load_b_frag(w0, CH, (kk + 1) * 32);
                b1n = load_b_frag(w1, CH, (kk + 1) * 32);
            }
            acc0 = wmma_bf16(a, b0c, acc0);
            acc1 = wmma_bf16(a, b1c, acc1);
            b0c = b0n; b1c = b1n;
        }
#pragma unroll
        for (int which = 0; which < 2; ++which) {
            const v8f acc = which ? acc1 : acc0;
            const int o = (which ? ob1 : ob0) + nl;
            if (o < 512) {
                int hd = o >> 7, d = o & 127;
                __bf16* dst = qbf + ((size_t)b * HEADS + hd) * LTOK * DHEAD;
#pragma unroll
                for (int r = 0; r < 8; ++r)
                    dst[(size_t)(mt * 16 + half * 8 + r) * DHEAD + d] = (__bf16)(acc[r] * SCALE);
            } else if (o < 1024) {
                int o2 = o - 512, hd = o2 >> 7, d = o2 & 127;
                __bf16* dst = kbf + ((size_t)b * HEADS + hd) * LTOK * DHEAD;
#pragma unroll
                for (int r = 0; r < 8; ++r)
                    dst[(size_t)(mt * 16 + half * 8 + r) * DHEAD + d] = (__bf16)acc[r];
            } else {
                int o2 = o - 1024, hd = o2 >> 7, d = o2 & 127;
                __bf16* dst = vt + (((size_t)b * HEADS + hd) * DHEAD + d) * LTOK;
#pragma unroll
                for (int r = 0; r < 8; ++r)
                    dst[mt * 16 + half * 8 + r] = (__bf16)acc[r];
            }
        }
    }
}

// ---------------- 4) q x rel GEMMs (WMMA) ----------------

__global__ void __launch_bounds__(128) k_qr(const __bf16* __restrict__ qbf,
                                            const __bf16* __restrict__ relw,
                                            const __bf16* __restrict__ relh,
                                            float* __restrict__ qrw,
                                            float* __restrict__ qrh) {
    const int hd = blockIdx.y, b = blockIdx.z;
    const int wave = threadIdx.x >> 5, lane = threadIdx.x & 31;
    const int half = lane >> 4, nl = lane & 15;
    const int mt = blockIdx.x * 4 + wave;
    const size_t ho = (size_t)b * HEADS + hd;

    const __bf16* qp = qbf + ho * LTOK * DHEAD + (size_t)mt * 16 * DHEAD;
    v16bf af[4];
#pragma unroll
    for (int kk = 0; kk < 4; ++kk) af[kk] = load_a_frag(qp, DHEAD, kk * 32);

    for (int which = 0; which < 2; ++which) {
        const __bf16* rel = which ? relh : relw;
        float* dst = (which ? qrh : qrw) + ho * LTOK * 64 + (size_t)mt * 16 * 64;
#pragma unroll
        for (int nt = 0; nt < 4; ++nt) {
            v8f acc = {};
#pragma unroll
            for (int kk = 0; kk < 4; ++kk) {
                v16bf bf = load_b_frag(rel + (size_t)nt * 16 * DHEAD, DHEAD, kk * 32);
                acc = wmma_bf16(af[kk], bf, acc);
            }
#pragma unroll
            for (int r = 0; r < 8; ++r)
                dst[(size_t)(half * 8 + r) * 64 + nt * 16 + nl] = acc[r];
        }
    }
}

// ---------------- 5) flash attention with TDM-staged K/V ----------------

__global__ void __launch_bounds__(128) k_attn(const __bf16* __restrict__ qbf,
                                              const __bf16* __restrict__ kbf,
                                              const __bf16* __restrict__ vt,
                                              const float* __restrict__ qrw,
                                              const float* __restrict__ qrh,
                                              float* __restrict__ attn) {
    const int qblk = blockIdx.x, hd = blockIdx.y, b = blockIdx.z;
    const int wave = threadIdx.x >> 5, lane = threadIdx.x & 31;
    const int half = lane >> 4, nl = lane & 15;
    const size_t ho = (size_t)b * HEADS + hd;
    const int qbase = qblk * 64;

    __shared__ float qw_s[64][64];      // 16 KB
    __shared__ float qh_s[64][64];      // 16 KB
    __shared__ __bf16 Kt[32][128];      //  8 KB
    __shared__ __bf16 Vs[128][32];      //  8 KB
    __shared__ __bf16 Pl[4][16][32];    //  4 KB

    const float* qrw_p = qrw + ho * LTOK * 64;
    const float* qrh_p = qrh + ho * LTOK * 64;
    for (int i = threadIdx.x; i < 64 * 64; i += 128) {
        int r = i >> 6, cc = i & 63;
        qw_s[r][cc] = qrw_p[(size_t)(qbase + r) * 64 + cc];
        qh_s[r][cc] = qrh_p[(size_t)(qbase + r) * 64 + cc];
    }

    const __bf16* qp = qbf + ho * LTOK * DHEAD + (size_t)(qbase + wave * 16) * DHEAD;
    const __bf16* kp = kbf + ho * LTOK * DHEAD;
    const __bf16* vp = vt + ho * (size_t)DHEAD * LTOK;

    v16bf qa[4];
#pragma unroll
    for (int kk = 0; kk < 4; ++kk) qa[kk] = load_a_frag(qp, DHEAD, kk * 32);

    v8f Oacc[8] = {};
    float mrun[8], lrun[8];
#pragma unroll
    for (int r = 0; r < 8; ++r) { mrun[r] = -1e30f; lrun[r] = 0.f; }

    const int iloc0 = wave * 16 + half * 8;

    for (int ch = 0; ch < 32; ++ch) {
        const int key0 = ch * 32;
        __syncthreads();

#if HAVE_TDM
        if (wave == 0) {
            tdm_load_2d((unsigned)(size_t)&Kt[0][0],
                        (unsigned long long)(size_t)(kp + (size_t)key0 * DHEAD),
                        DHEAD, LTOK, DHEAD, DHEAD, 32);
            tdm_load_2d((unsigned)(size_t)&Vs[0][0],
                        (unsigned long long)(size_t)(vp + key0),
                        LTOK, DHEAD, LTOK, 32, DHEAD);
            tdm_wait0();
        }
#else
        for (int i = threadIdx.x; i < 512; i += 128)
            ((v8bf*)&Kt[0][0])[i] = *(const v8bf*)(kp + (size_t)key0 * DHEAD + i * 8);
        for (int i = threadIdx.x; i < 512; i += 128) {
            int d = i >> 2, seg = i & 3;
            ((v8bf*)&Vs[d][0])[seg] = *(const v8bf*)(vp + (size_t)d * LTOK + key0 + seg * 8);
        }
#endif
        __syncthreads();

        v8f s0 = {}, s1 = {};
#pragma unroll
        for (int kk = 0; kk < 4; ++kk) {
            v16bf b0 = load_b_frag(&Kt[0][0], DHEAD, kk * 32);
            v16bf b1 = load_b_frag(&Kt[16][0], DHEAD, kk * 32);
            s0 = wmma_bf16(qa[kk], b0, s0);
            s1 = wmma_bf16(qa[kk], b1, s1);
        }

        const int j0 = key0 + nl, j1 = key0 + 16 + nl;
        const int w2a = j0 & 31, h2a = j0 >> 5;
        const int w2b = j1 & 31, h2b = j1 >> 5;
        float alpha[8];
#pragma unroll
        for (int r = 0; r < 8; ++r) {
            int il = iloc0 + r;
            int qi = qbase + il;
            int w1 = qi & 31, h1 = qi >> 5;
            s0[r] += qw_s[il][w2a - w1 + 31] + qh_s[il][h2a - h1 + 31];
            s1[r] += qw_s[il][w2b - w1 + 31] + qh_s[il][h2b - h1 + 31];
            float mx = fmaxf(s0[r], s1[r]);
            mx = fmaxf(mx, __shfl_xor(mx, 1, 16));
            mx = fmaxf(mx, __shfl_xor(mx, 2, 16));
            mx = fmaxf(mx, __shfl_xor(mx, 4, 16));
            mx = fmaxf(mx, __shfl_xor(mx, 8, 16));
            float mnew = fmaxf(mrun[r], mx);
            alpha[r] = __expf(mrun[r] - mnew);
            mrun[r] = mnew;
            float p0 = __expf(s0[r] - mnew);
            float p1 = __expf(s1[r] - mnew);
            s0[r] = p0; s1[r] = p1;
            float rs = p0 + p1;
            rs += __shfl_xor(rs, 1, 16);
            rs += __shfl_xor(rs, 2, 16);
            rs += __shfl_xor(rs, 4, 16);
            rs += __shfl_xor(rs, 8, 16);
            lrun[r] = lrun[r] * alpha[r] + rs;
        }
#pragma unroll
        for (int t = 0; t < 8; ++t)
#pragma unroll
            for (int r = 0; r < 8; ++r) Oacc[t][r] *= alpha[r];

#pragma unroll
        for (int r = 0; r < 8; ++r) {
            Pl[wave][half * 8 + r][nl]      = (__bf16)s0[r];
            Pl[wave][half * 8 + r][16 + nl] = (__bf16)s1[r];
        }
        asm volatile("s_wait_dscnt 0x0" ::: "memory");
        v16bf pa = load_a_frag(&Pl[wave][0][0], 32, 0);

#pragma unroll
        for (int t = 0; t < 8; ++t) {
            v16bf vb = load_b_frag(&Vs[t * 16][0], 32, 0);
            Oacc[t] = wmma_bf16(pa, vb, Oacc[t]);
        }
    }

    float* op = attn + ho * LTOK * DHEAD + (size_t)(qbase + wave * 16) * DHEAD;
#pragma unroll
    for (int t = 0; t < 8; ++t)
#pragma unroll
        for (int r = 0; r < 8; ++r) {
            int m = half * 8 + r;
            op[(size_t)m * DHEAD + t * 16 + nl] = Oacc[t][r] / lrun[r];
        }
}

// ---------------- 6) residual + BN2 + ReLU ----------------

__global__ void k_resid_bn2(const float* __restrict__ x, const float* __restrict__ attn,
                            const float* __restrict__ g, const float* __restrict__ be,
                            const float* __restrict__ mu, const float* __restrict__ va,
                            float* __restrict__ x2, __bf16* __restrict__ r2) {
    int idx = blockIdx.x * blockDim.x + threadIdx.x;
    if (idx >= BATCH * CH * LTOK) return;
    int t = idx % LTOK;
    int c = (idx / LTOK) % CH;
    int b = idx / (CH * LTOK);
    int hd = c >> 7, d = c & 127;
    float s = x[idx] + attn[(((size_t)b * HEADS + hd) * LTOK + t) * DHEAD + d];
    x2[idx] = s;
    float inv = g[c] * __frsqrt_rn(va[c] + EPS);
    float r = fmaxf(s * inv + (be[c] - mu[c] * inv), 0.f);
    r2[((size_t)b * LTOK + t) * CH + c] = (__bf16)r;
}

// ---------------- 7) 3x3 conv: LDS halo + implicit GEMM (WMMA) --------------
// grid (64, 8), block 128 ; halo 3x18x512 bf16 shared by the 4 waves.

__global__ void __launch_bounds__(128) k_conv(const __bf16* __restrict__ r2,
                                              const __bf16* __restrict__ wfcT,
                                              const float* __restrict__ bias,
                                              const float* __restrict__ x2,
                                              float* __restrict__ out) {
    const int mt = blockIdx.x, b = blockIdx.y;
    const int wave = threadIdx.x >> 5, lane = threadIdx.x & 31;
    const int half = lane >> 4, nl = lane & 15;
    const int h = mt >> 1, wbase = (mt & 1) * 16;   // tile = row h, cols wbase..wbase+15

    __shared__ __bf16 halo[3][18][CH];              // 54 KB

    // cooperative halo stage: zero-fill out-of-image chunks
    for (int i = threadIdx.x; i < 3 * 18 * (CH / 8); i += 128) {
        int seg = i & 63;
        int col = (i >> 6) % 18;
        int r3 = i / (18 * 64);
        int hr = h - 1 + r3, wc = wbase - 1 + col;
        v8bf val = {};
        if (hr >= 0 && hr < HHDIM && wc >= 0 && wc < WWDIM)
            val = *(const v8bf*)(r2 + (((size_t)b * HHDIM + hr) * WWDIM + wc) * CH + seg * 8);
        *((v8bf*)&halo[r3][col][0] + seg) = val;
    }
    __syncthreads();

    v8f acc[8] = {};
    for (int tap = 0; tap < 9; ++tap) {
        const int dy = tap / 3, dx = tap % 3;
        const __bf16* arow = &halo[dy][(lane & 15) + dx][0];   // per-lane A row (LDS)
        const __bf16* wt = wfcT + (size_t)tap * CH * CH + (size_t)(wave * 128) * CH;
        if (tap < 8) __builtin_prefetch(wfcT + (size_t)(tap + 1) * CH * CH + (size_t)(wave * 128) * CH, 0, 1);

        v16bf acur = load_a_frag_rowptr(arow, 0);
#pragma unroll 4
        for (int kk = 0; kk < 16; ++kk) {
            v16bf anxt = acur;
            if (kk < 15) anxt = load_a_frag_rowptr(arow, (kk + 1) * 32);
            v16bf bcur = load_b_frag(wt, CH, kk * 32);
#pragma unroll
            for (int nt = 0; nt < 8; ++nt) {
                v16bf bnxt = bcur;
                if (nt < 7) bnxt = load_b_frag(wt + (size_t)(nt + 1) * 16 * CH, CH, kk * 32);
                acc[nt] = wmma_bf16(acur, bcur, acc[nt]);
                bcur = bnxt;
            }
            acur = anxt;
        }
    }
#pragma unroll
    for (int nt = 0; nt < 8; ++nt) {
        int o = wave * 128 + nt * 16 + nl;
        float bv = bias[o];
#pragma unroll
        for (int r = 0; r < 8; ++r) {
            size_t idx = ((size_t)b * CH + o) * LTOK + mt * 16 + half * 8 + r;
            out[idx] = acc[nt][r] + bv + x2[idx];
        }
    }
}

// ---------------------------------------------------------------------------

extern "C" void kernel_launch(void* const* d_in, const int* in_sizes, int n_in,
                              void* d_out, int out_size, void* d_ws, size_t ws_size,
                              hipStream_t stream) {
    const float* x     = (const float*)d_in[0];
    const float* w_qk  = (const float*)d_in[1];
    const float* w_v   = (const float*)d_in[2];
    const float* rel_h = (const float*)d_in[3];
    const float* rel_w = (const float*)d_in[4];
    const float* g1 = (const float*)d_in[5];
    const float* b1 = (const float*)d_in[6];
    const float* m1 = (const float*)d_in[7];
    const float* v1 = (const float*)d_in[8];
    const float* g2 = (const float*)d_in[9];
    const float* b2 = (const float*)d_in[10];
    const float* m2 = (const float*)d_in[11];
    const float* v2 = (const float*)d_in[12];
    const float* w_fc    = (const float*)d_in[13];
    const float* bias_fc = (const float*)d_in[14];
    float* out = (float*)d_out;

    char* ws = (char*)d_ws;
    __bf16* xb    = (__bf16*)(ws + 0);            //  8 MB [b][t][c]
    __bf16* wqkb  = (__bf16*)(ws + 8388608);      //  1 MB
    __bf16* wvb   = (__bf16*)(ws + 9437184);      //  0.5 MB
    __bf16* relwp = (__bf16*)(ws + 9961472);      //  16 KB (64x128)
    __bf16* relhp = (__bf16*)(ws + 9977856);      //  16 KB
    __bf16* wfcT  = (__bf16*)(ws + 9994240);      //  4.5 MB [9][O][C]
    __bf16* qbf   = (__bf16*)(ws + 14712832);     //  8 MB [b][h][t][d]
    __bf16* kbf   = (__bf16*)(ws + 23101440);     //  8 MB
    __bf16* vt    = (__bf16*)(ws + 31490048);     //  8 MB [b][h][d][t]
    float*  qrw   = (float*)(ws + 39878656);      //  8 MB [b][h][t][64]
    float*  qrh   = (float*)(ws + 48267264);      //  8 MB
    float*  attn  = (float*)(ws + 56655872);      // 16 MB [b][h][t][d]
    float*  x2    = (float*)(ws + 73433088);      // 16 MB NCHW
    __bf16* r2    = (__bf16*)(ws + 90210304);     //  8 MB [b][t][c]

    const int NE = BATCH * CH * LTOK;             // 4194304

    k_bn1_pack<<<(NE + 255) / 256, 256, 0, stream>>>(x, g1, b1, m1, v1, xb);
    k_cvt_bf16<<<(1024 * 512 + 255) / 256, 256, 0, stream>>>(w_qk, wqkb, 1024 * 512);
    k_cvt_bf16<<<(512 * 512 + 255) / 256, 256, 0, stream>>>(w_v, wvb, 512 * 512);
    k_cvt_rel<<<(64 * 128 + 255) / 256, 256, 0, stream>>>(rel_h, rel_w, relhp, relwp);
    k_cvt_wfc<<<(9 * 512 * 512 + 255) / 256, 256, 0, stream>>>(w_fc, wfcT);

    k_qkv<<<dim3(64, BATCH), 128, 0, stream>>>(xb, wqkb, wvb, qbf, kbf, vt);
    k_qr<<<dim3(16, HEADS, BATCH), 128, 0, stream>>>(qbf, relwp, relhp, qrw, qrh);
    k_attn<<<dim3(16, HEADS, BATCH), 128, 0, stream>>>(qbf, kbf, vt, qrw, qrh, attn);
    k_resid_bn2<<<(NE + 255) / 256, 256, 0, stream>>>(x, attn, g2, b2, m2, v2, x2, r2);
    k_conv<<<dim3(64, BATCH), 128, 0, stream>>>(r2, wfcT, bias_fc, x2, out);
    (void)in_sizes; (void)n_in; (void)out_size; (void)ws_size;
}
